// MSMCM_23828478558391
// MI455X (gfx1250) — matrix operand order repaired and checked by reading, not compiled, not run on hardware
//
#include <hip/hip_runtime.h>
#include <hip/hip_bf16.h>

// Problem constants (match reference)
#define BB      4
#define CC      32
#define HH      192
#define WWID    192
#define HWSZ    (HH*WWID)          // 36864
#define NBR     4
#define BETA_F  15.0f
#define INVBETA (1.0f/15.0f)
#define EPS_F   1e-5f
#define NPERCH  (BB*HWSZ)          // 147456 (BN count over B,H,W)
#define NTOT    ((size_t)NBR*BB*CC*HWSZ)   // 18,874,368 floats per morph buffer
#define NOUT    ((size_t)BB*CC*HWSZ)       // 4,718,592 output elements

typedef unsigned int  u32;
typedef unsigned short u16;
typedef __attribute__((ext_vector_type(16))) __bf16 v16bf;
typedef __attribute__((ext_vector_type(8)))  float  v8f;
typedef __attribute__((ext_vector_type(4)))  u32    u32x4;

union Frag { v16bf v; u32x4 q[2]; };

__device__ __forceinline__ u16 f2bf(float f) {
    union { float f; u32 u; } a; a.f = f;
    u32 r = a.u + 0x7FFFu + ((a.u >> 16) & 1u);   // round-to-nearest-even
    return (u16)(r >> 16);
}

// ---------------- zero the stats scratch ----------------
__global__ void k_zero(float* p, int n) {
    int i = blockIdx.x * blockDim.x + threadIdx.x;
    if (i < n) p[i] = 0.0f;
}

// ---------------- soft dilate: y = logsumexp(beta*(x_tap + w))/beta ----------------
__global__ __launch_bounds__(256) void k_dilate(const float* __restrict__ x,
                                                const float* __restrict__ wd,
                                                float* __restrict__ y) {
    const int CHUNKS = HWSZ / 256;               // 144
    int blk   = blockIdx.x;
    int chunk = blk % CHUNKS;
    int ch    = blk / CHUNKS;                    // i*B*C + b*C + c  (ib*C + c)
    int c  = ch % CC;
    int ib = ch / CC;                            // i*B + b
    int i  = ib / BB;
    int b  = ib % BB;
    int d  = i + 1;

    int hw = chunk * 256 + threadIdx.x;
    int h = hw / WWID, w = hw % WWID;

    const float* wp = wd + (i*CC + c)*9;
    const float* xc = x  + ((size_t)(b*CC + c))*HWSZ;

    float v[9];
#pragma unroll
    for (int t = 0; t < 9; ++t) {
        int ti = t / 3, tj = t % 3;
        int hh = h + (ti - 1)*d;
        int ww = w + (tj - 1)*d;
        float xv = (hh >= 0 && hh < HH && ww >= 0 && ww < WWID) ? xc[hh*WWID + ww] : 0.0f;
        v[t] = xv + wp[t];
    }
    float m = v[0];
#pragma unroll
    for (int t = 1; t < 9; ++t) m = fmaxf(m, v[t]);
    float s = 0.0f;
#pragma unroll
    for (int t = 0; t < 9; ++t) s += __expf(BETA_F * (v[t] - m));
    y[(size_t)ch * HWSZ + hw] = m + __logf(s) * INVBETA;
}

// ---------------- soft erode + BN1 stats: y2 = -logsumexp(beta*(w - y_tap))/beta ----------------
__global__ __launch_bounds__(256) void k_erode(const float* __restrict__ y1,
                                               const float* __restrict__ we,
                                               float* __restrict__ y2,
                                               float* __restrict__ sum1,
                                               float* __restrict__ sq1) {
    const int CHUNKS = HWSZ / 256;
    int blk   = blockIdx.x;
    int chunk = blk % CHUNKS;
    int ch    = blk / CHUNKS;                    // ib*C + c
    int c  = ch % CC;
    int ib = ch / CC;
    int i  = ib / BB;
    int d  = i + 1;

    int tid = threadIdx.x;
    int hw  = chunk * 256 + tid;
    int h = hw / WWID, w = hw % WWID;

    const float* wp = we + (i*CC + c)*9;
    const float* yc = y1 + (size_t)ch * HWSZ;

    float v[9];
#pragma unroll
    for (int t = 0; t < 9; ++t) {
        int ti = t / 3, tj = t % 3;
        int hh = h + (ti - 1)*d;
        int ww = w + (tj - 1)*d;
        float yv = (hh >= 0 && hh < HH && ww >= 0 && ww < WWID) ? yc[hh*WWID + ww] : 0.0f;
        v[t] = wp[t] - yv;
    }
    float m = v[0];
#pragma unroll
    for (int t = 1; t < 9; ++t) m = fmaxf(m, v[t]);
    float s = 0.0f;
#pragma unroll
    for (int t = 0; t < 9; ++t) s += __expf(BETA_F * (v[t] - m));
    float r = -(m + __logf(s) * INVBETA);
    y2[(size_t)ch * HWSZ + hw] = r;

    // block reduction of sum / sumsq for this (branch,channel)
    __shared__ float s1[256];
    __shared__ float s2[256];
    s1[tid] = r; s2[tid] = r * r;
    __syncthreads();
#pragma unroll
    for (int off = 128; off > 0; off >>= 1) {
        if (tid < off) { s1[tid] += s1[tid + off]; s2[tid] += s2[tid + off]; }
        __syncthreads();
    }
    if (tid == 0) {
        int k = i*CC + c;
        atomicAdd(&sum1[k], s1[0]);
        atomicAdd(&sq1[k],  s2[0]);
    }
}

// ---------------- fold BN1 into scale/bias: yhat = y*A1 + B1 ----------------
__global__ void k_stats1(const float* sum1, const float* sq1,
                         const float* gs, const float* bs,
                         float* A1, float* B1) {
    int k = threadIdx.x;                         // 0..127 = i*C + c
    float inv = 1.0f / (float)NPERCH;
    float mu  = sum1[k] * inv;
    float var = sq1[k] * inv - mu * mu;
    float rs  = rsqrtf(var + EPS_F);
    float a   = rs * gs[k];
    A1[k] = a;
    B1[k] = bs[k] - mu * a;
}

// ---------------- 1x1 conv GEMM via WMMA bf16 (M=147456, K=128, N=32) + BN2 stats ----------------
__global__ __launch_bounds__(256) void k_gemm(const float* __restrict__ y2,
                                              const float* __restrict__ A1,
                                              const float* __restrict__ B1,
                                              const float* __restrict__ Wc,
                                              float* __restrict__ z,
                                              float* __restrict__ sum2,
                                              float* __restrict__ sq2) {
    __shared__ u16 aT[128 * 136];                // 128 positions x 128 ch, +8 pad (bank-conflict free)
    __shared__ u16 wT[32 * 136];                 // 32 out-ch x 128 in-ch, +8 pad

    int tid = threadIdx.x;
    int p0  = blockIdx.x * 128;                  // HWSZ % 128 == 0 -> never crosses batch boundary
    int b   = p0 / HWSZ;
    int hw0 = p0 % HWSZ;

    // stage Wc as bf16
    for (int idx = tid; idx < 32 * 128; idx += 256) {
        int o = idx >> 7, ii = idx & 127;
        wT[o * 136 + ii] = f2bf(Wc[idx]);
    }

    // stage activation tile: BN1 + ReLU + bf16 convert, coalesced per channel row
    int pos = tid & 127;
    int kst = tid >> 7;                          // 0 or 1
    for (int k = kst; k < 128; k += 2) {
        int i = k >> 5, c = k & 31;
        const float* src = y2 + ((size_t)((i*BB + b)*CC + c))*HWSZ + hw0;
        if (k + 2 < 128) {
            int kn = k + 2;
            const float* srcn = y2 + ((size_t)(((kn >> 5)*BB + b)*CC + (kn & 31)))*HWSZ + hw0;
            __builtin_prefetch(srcn + pos, 0, 0);    // global_prefetch_b8
        }
        float t = fmaxf(src[pos] * A1[k] + B1[k], 0.0f);
        aT[pos * 136 + k] = f2bf(t);
    }
    __syncthreads();

    int wave = tid >> 5, lane = tid & 31;
    int mrow = lane & 15, half = lane >> 4;      // half: K-group / M-half selector per ISA layout
    int arow = wave * 16 + mrow;
    int n0   = lane & 15;

    v8f acc0 = {}; v8f acc1 = {};
#pragma unroll
    for (int kc = 0; kc < 4; ++kc) {
        int kb = kc * 32;
        Frag fa, fb0, fb1;
        // A 16x32 bf16: lane holds row M; VGPR0-3 = K[half*8 .. +7], VGPR4-7 = K[16+half*8 .. +7]
        fa.q[0] = *(const u32x4*)&aT[arow * 136 + kb + half * 8];
        fa.q[1] = *(const u32x4*)&aT[arow * 136 + kb + 16 + half * 8];
        // B 32x16 bf16: lane holds column N; VGPR0-3 = K[half*8 .. +7], VGPR4-7 = K[16+half*8 .. +7]
        const u16* w0 = &wT[n0 * 136 + kb + half * 8];
        fb0.q[0] = *(const u32x4*)(w0);
        fb0.q[1] = *(const u32x4*)(w0 + 16);
        const u16* w1 = &wT[(n0 + 16) * 136 + kb + half * 8];
        fb1.q[0] = *(const u32x4*)(w1);
        fb1.q[1] = *(const u32x4*)(w1 + 16);
        acc0 = __builtin_amdgcn_wmma_f32_16x16x32_bf16(false, fa.v, false, fb0.v,
                                                       (short)0, acc0, false, false);
        acc1 = __builtin_amdgcn_wmma_f32_16x16x32_bf16(false, fa.v, false, fb1.v,
                                                       (short)0, acc1, false, false);
    }

    // D layout: VGPR r -> M = r + 8*half ; N = (lane&15) (+16 for second tile)
    float s0 = 0.f, ss0 = 0.f, s1v = 0.f, ss1v = 0.f;
#pragma unroll
    for (int r = 0; r < 8; ++r) {
        int m = r + 8 * half;
        int p = wave * 16 + m;
        float v0 = acc0[r];
        float v1 = acc1[r];
        z[((size_t)(b*CC + n0     ))*HWSZ + hw0 + p] = v0;
        z[((size_t)(b*CC + n0 + 16))*HWSZ + hw0 + p] = v1;
        s0 += v0; ss0 += v0 * v0; s1v += v1; ss1v += v1 * v1;
    }
    atomicAdd(&sum2[n0],      s0);
    atomicAdd(&sq2[n0],       ss0);
    atomicAdd(&sum2[n0 + 16], s1v);
    atomicAdd(&sq2[n0 + 16],  ss1v);
}

// ---------------- fold BN2 ----------------
__global__ void k_stats2(const float* sum2, const float* sq2,
                         const float* g, const float* bb,
                         float* A2, float* B2) {
    int o = threadIdx.x;                         // 0..31
    float inv = 1.0f / (float)NPERCH;
    float mu  = sum2[o] * inv;
    float var = sq2[o] * inv - mu * mu;
    float rs  = rsqrtf(var + EPS_F);
    float a   = rs * g[o];
    A2[o] = a;
    B2[o] = bb[o] - mu * a;
}

// ---------------- final BN2 + ReLU ----------------
__global__ __launch_bounds__(256) void k_final(const float* __restrict__ z,
                                               const float* __restrict__ A2,
                                               const float* __restrict__ B2,
                                               float* __restrict__ out) {
    size_t idx = (size_t)blockIdx.x * 256 + threadIdx.x;
    if (idx >= NOUT) return;
    int o = (int)((idx / HWSZ) % CC);
    out[idx] = fmaxf(z[idx] * A2[o] + B2[o], 0.0f);
}

extern "C" void kernel_launch(void* const* d_in, const int* in_sizes, int n_in,
                              void* d_out, int out_size, void* d_ws, size_t ws_size,
                              hipStream_t stream) {
    (void)in_sizes; (void)n_in; (void)out_size; (void)ws_size;
    const float* x  = (const float*)d_in[0];
    const float* wd = (const float*)d_in[1];
    const float* we = (const float*)d_in[2];
    const float* gs = (const float*)d_in[3];
    const float* bs = (const float*)d_in[4];
    const float* Wc = (const float*)d_in[5];
    const float* g  = (const float*)d_in[6];
    const float* bb = (const float*)d_in[7];
    float* out = (float*)d_out;

    float* ws   = (float*)d_ws;
    float* bufA = ws;                            // NTOT floats (dilated; later reused as z)
    float* bufB = ws + NTOT;                     // NTOT floats (eroded)
    float* st   = ws + 2 * NTOT;
    float* sum1 = st;                            // 128
    float* sq1  = st + 128;
    float* A1   = st + 256;                      // 128
    float* B1   = st + 384;                      // 128
    float* sum2 = st + 512;                      // 32
    float* sq2  = st + 544;                      // 32
    float* A2   = st + 576;                      // 32
    float* B2   = st + 608;                      // 32
    float* z    = bufA;                          // reuse (NOUT <= NTOT)

    const int CHUNKS = HWSZ / 256;                         // 144
    const int morphBlocks = NBR * BB * CC * CHUNKS;        // 73728

    k_zero<<<3, 256, 0, stream>>>(st, 640);
    k_dilate<<<morphBlocks, 256, 0, stream>>>(x, wd, bufA);
    k_erode <<<morphBlocks, 256, 0, stream>>>(bufA, we, bufB, sum1, sq1);
    k_stats1<<<1, 128, 0, stream>>>(sum1, sq1, gs, bs, A1, B1);
    k_gemm  <<<(int)(NOUT / CC / 128), 256, 0, stream>>>(bufB, A1, B1, Wc, z, sum2, sq2);
    k_stats2<<<1, 32, 0, stream>>>(sum2, sq2, g, bb, A2, B2);
    k_final <<<(int)((NOUT + 255) / 256), 256, 0, stream>>>(z, A2, B2, out);
}
